// GraphormerLayer_84894323573344
// MI455X (gfx1250) — compile-verified, matrix operand order checked by reading, not run on hardware
//
#include <hip/hip_runtime.h>
#include <hip/hip_bf16.h>

typedef __attribute__((ext_vector_type(16))) __bf16 v16bf;
typedef __attribute__((ext_vector_type(8)))  float  v8f;

#define WMMA_BF16(a, b, c) \
  __builtin_amdgcn_wmma_f32_16x16x32_bf16(false, (a), false, (b), (short)0, (c), false, false)

static __device__ __forceinline__ v8f zero8() {
  v8f z = {0.f, 0.f, 0.f, 0.f, 0.f, 0.f, 0.f, 0.f};
  return z;
}

// A-fragment (and symmetric B-fragment from a pre-transposed [N][K] array):
// 16-bit 16x32 layout per CDNA5 ISA 7.12.2:
//   lane: row = row0 + lane%16 ; K base = (lane<16 ? 0 : 8)
//   VGPR 0..3 : K = base+0..7 (pairs) ; VGPR 4..7 : K = base+16..23 (pairs)
static __device__ __forceinline__ v16bf load_frag_bf16(const __bf16* __restrict__ P,
                                                       int ld, int row0, int k0) {
  const int lane = threadIdx.x & 31;
  const int r  = row0 + (lane & 15);
  const int kb = k0 + ((lane & 16) ? 8 : 0);
  const __bf16* p = P + (size_t)r * ld + kb;
  v16bf f;
#pragma unroll
  for (int i = 0; i < 4; ++i) { f[2*i]   = p[2*i];      f[2*i+1]   = p[2*i+1]; }
#pragma unroll
  for (int i = 0; i < 4; ++i) { f[8+2*i] = p[16+2*i];   f[8+2*i+1] = p[16+2*i+1]; }
  return f;
}

static __device__ __forceinline__ v16bf load_frag_f32(const float* __restrict__ P,
                                                      int ld, int row0, int k0) {
  const int lane = threadIdx.x & 31;
  const int r  = row0 + (lane & 15);
  const int kb = k0 + ((lane & 16) ? 8 : 0);
  const float* p = P + (size_t)r * ld + kb;
  v16bf f;
#pragma unroll
  for (int i = 0; i < 4; ++i) { f[2*i]   = (__bf16)p[2*i];    f[2*i+1]   = (__bf16)p[2*i+1]; }
#pragma unroll
  for (int i = 0; i < 4; ++i) { f[8+2*i] = (__bf16)p[16+2*i]; f[8+2*i+1] = (__bf16)p[16+2*i+1]; }
  return f;
}

// ---------------------------------------------------------------- utilities
__global__ void zero_f32(float* __restrict__ p, size_t n) {
  size_t i = (size_t)blockIdx.x * blockDim.x + threadIdx.x;
  if (i < n) p[i] = 0.f;
}
__global__ void zero_i32(int* __restrict__ p, size_t n) {
  size_t i = (size_t)blockIdx.x * blockDim.x + threadIdx.x;
  if (i < n) p[i] = 0;
}
__global__ void deg_count(const int* __restrict__ src, int* __restrict__ deg, int E) {
  int e = blockIdx.x * blockDim.x + threadIdx.x;
  if (e < E) atomicAdd(&deg[src[e]], 1);
}
// x = node_feature + centrality[clip(deg,0,127)], packed to bf16
__global__ void node_prep(const float* __restrict__ nf, const float* __restrict__ cen,
                          const int* __restrict__ deg, __bf16* __restrict__ x_bf,
                          size_t total) {
  size_t i = (size_t)blockIdx.x * blockDim.x + threadIdx.x;
  if (i >= total) return;
  int t = (int)(i >> 9);
  int c = (int)(i & 511);
  int d = deg[t];
  d = d < 0 ? 0 : (d > 127 ? 127 : d);
  x_bf[i] = (__bf16)(nf[i] + cen[(size_t)d * 512 + c]);
}
__global__ void inv_deg_k(const int* __restrict__ deg, float* __restrict__ invs, int T) {
  int t = blockIdx.x * blockDim.x + threadIdx.x;
  if (t >= T) return;
  float d = (float)deg[t];
  if (d < 1.f) d = 1.f;
  invs[t] = rsqrtf(d);
}
// dst[n*K+k] = (bf16)src[k*ld + c0 + n]   (weight -> bf16 transposed [N][K])
__global__ void cvt_transpose(const float* __restrict__ src, __bf16* __restrict__ dst,
                              int K, int Nc, int c0, int ld, size_t total) {
  size_t i = (size_t)blockIdx.x * blockDim.x + threadIdx.x;
  if (i >= total) return;
  int n = (int)(i / K);
  int k = (int)(i % K);
  dst[(size_t)n * K + k] = (__bf16)src[(size_t)k * ld + c0 + n];
}
// vT[g][h][d][n] = v_f[(g*128+n)*512 + h*64 + d]
__global__ void pack_vT(const float* __restrict__ v_f, __bf16* __restrict__ vT, size_t total) {
  size_t i = (size_t)blockIdx.x * blockDim.x + threadIdx.x;
  if (i >= total) return;
  int t = (int)(i >> 9);
  int c = (int)(i & 511);
  int g = t >> 7, n = t & 127;
  int h = c >> 6, d = c & 63;
  vT[(((size_t)(g * 8 + h) * 64) + d) * 128 + n] = (__bf16)v_f[i];
}
// h0 = dense_pred + (scat + deg*b_e) * inv_sqrt_deg, packed to bf16
__global__ void combine_k(const float* __restrict__ dense, const float* __restrict__ scat,
                          const int* __restrict__ deg, const float* __restrict__ invs,
                          const float* __restrict__ b_e, __bf16* __restrict__ h0,
                          size_t total) {
  size_t i = (size_t)blockIdx.x * blockDim.x + threadIdx.x;
  if (i >= total) return;
  int t = (int)(i >> 9);
  int c = (int)(i & 511);
  float eo = scat[i] + (float)deg[t] * b_e[c];
  h0[i] = (__bf16)(dense[i] + eo * invs[t]);
}

// ------------------------------------------------------- generic WMMA GEMM
// Block = 8 waves -> 8 M-tiles (128 rows) x one 64-col N-strip.
// Shared 64xK weight strip is staged into LDS with async global->LDS copies
// (ASYNCcnt), double-buffered in 128-K chunks overlapping the WMMA work.
// LDS row stride 136 bf16 (68 dwords) -> 4-bank skew, conflict-free reads.
#define BSTRIDE 136

// one 64-row x 128-K bf16 chunk: 1024 b128 async loads, 4 per thread
static __device__ __forceinline__ void stage_b_chunk(const __bf16* __restrict__ Bt, int ldb,
                                                     int nbase, int k0,
                                                     __bf16* __restrict__ buf) {
  const int t = threadIdx.x;  // 0..255
#pragma unroll
  for (int j = 0; j < 4; ++j) {
    int i = t + (j << 8);     // 0..1023
    int row = i >> 4;         // 0..63
    int seg = i & 15;         // 16B segment within the 256B row
    const __bf16* g = Bt + (size_t)(nbase + row) * ldb + k0 + seg * 8;
    unsigned int lds_off = (unsigned int)(uintptr_t)(buf + row * BSTRIDE + seg * 8);
    asm volatile("global_load_async_to_lds_b128 %0, %1, off"
                 :: "v"(lds_off), "v"((unsigned long long)(uintptr_t)g)
                 : "memory");
  }
}

// epi: 0 = f32 out (+bias if non-null); 1 = bf16 out; 2 = bf16 out +bias +PReLU
__global__ void __launch_bounds__(256)
gemm_bf16_kernel(const __bf16* __restrict__ A, int lda,
                 const __bf16* __restrict__ Bt, int ldb,
                 int M, int N, int K,
                 const float* __restrict__ bias,
                 const float* __restrict__ prelu_p,
                 float* __restrict__ Cf, __bf16* __restrict__ Cb, int epi) {
  __shared__ __bf16 bstage[2][64 * BSTRIDE];
  const int lane = threadIdx.x & 31;
  const int w    = threadIdx.x >> 5;
  const int nstrips = N >> 6;
  const int mg = blockIdx.x / nstrips;
  const int ns = blockIdx.x % nstrips;
  const int mt = mg * 8 + w;          // this wave's 16-row M tile
  const int nbase = ns * 64;

  v8f acc[4];
#pragma unroll
  for (int j = 0; j < 4; ++j) acc[j] = zero8();

  stage_b_chunk(Bt, ldb, nbase, 0, bstage[0]);
  asm volatile("s_wait_asynccnt 0x0" ::: "memory");
  __syncthreads();

  const int nkc = K >> 7;             // 128-wide K chunks (K multiple of 128)
  for (int c = 0; c < nkc; ++c) {
    if (c + 1 < nkc)
      stage_b_chunk(Bt, ldb, nbase, (c + 1) << 7, bstage[(c + 1) & 1]);
    const __bf16* bb = bstage[c & 1];
#pragma unroll
    for (int kk = 0; kk < 4; ++kk) {
      v16bf a = load_frag_bf16(A, lda, mt * 16, (c << 7) + kk * 32);
#pragma unroll
      for (int j = 0; j < 4; ++j) {
        v16bf b = load_frag_bf16(bb, BSTRIDE, j * 16, kk * 32);
        acc[j] = WMMA_BF16(a, b, acc[j]);
      }
    }
    if (c + 1 < nkc) {
      asm volatile("s_wait_asynccnt 0x0" ::: "memory");
      __syncthreads();                // async chunk landed + buffer reads done
    }
  }

  const int rbase = (lane & 16) ? 8 : 0;
  const int cl = lane & 15;
  float pa = (epi == 2) ? prelu_p[0] : 0.f;
#pragma unroll
  for (int j = 0; j < 4; ++j) {
    int col = nbase + j * 16 + cl;
    float bv = bias ? bias[col] : 0.f;
#pragma unroll
    for (int r = 0; r < 8; ++r) {
      int row = mt * 16 + r + rbase;
      float v = acc[j][r] + bv;
      if (epi == 2) v = (v >= 0.f) ? v : v * pa;
      if (epi >= 1) Cb[(size_t)row * N + col] = (__bf16)v;
      else          Cf[(size_t)row * N + col] = v;
    }
  }
}

// -------------------------------------------------- fused attention (g, h)
// block = 256 threads = 8 waves, wave w handles query rows w*16..w*16+15
__global__ void __launch_bounds__(256)
attn_kernel(const __bf16* __restrict__ q_bf, const __bf16* __restrict__ k_bf,
            const __bf16* __restrict__ vT_bf, const int* __restrict__ dist_indices,
            const float* __restrict__ dist_bias, __bf16* __restrict__ attnout) {
  __shared__ __bf16 lds[8][16][136];   // 136 stride -> 4-bank skew per row
  const int lane = threadIdx.x & 31;
  const int w = threadIdx.x >> 5;
  const int g = blockIdx.x >> 3;
  const int h = blockIdx.x & 7;
  const int m0 = w * 16;

  const __bf16* qg = q_bf + ((size_t)g * 128) * 512 + h * 64;
  const __bf16* kg = k_bf + ((size_t)g * 128) * 512 + h * 64;

  // sim = q k^T   (K = DH = 64 -> two 32-chunks)
  v16bf a0 = load_frag_bf16(qg, 512, m0, 0);
  v16bf a1 = load_frag_bf16(qg, 512, m0, 32);
  v8f s[8];
#pragma unroll
  for (int j = 0; j < 8; ++j) {
    v16bf b0 = load_frag_bf16(kg, 512, j * 16, 0);
    v16bf b1 = load_frag_bf16(kg, 512, j * 16, 32);
    v8f acc = zero8();
    acc = WMMA_BF16(a0, b0, acc);
    acc = WMMA_BF16(a1, b1, acc);
    s[j] = acc;
  }
  const int rbase = (lane & 16) ? 8 : 0;
  const int cl = lane & 15;
  // scale + distance-bias gather (C-fragment element (m, key))
#pragma unroll
  for (int j = 0; j < 8; ++j) {
    int key = j * 16 + cl;
#pragma unroll
    for (int r = 0; r < 8; ++r) {
      int mg = m0 + r + rbase;
      int di = dist_indices[((size_t)g * 128 + mg) * 128 + key];
      s[j][r] = s[j][r] * 0.125f + dist_bias[di * 8 + h];
    }
  }
  // softmax over 128 keys: 8 regs per lane x 16 lanes of one half-wave == one row
#pragma unroll
  for (int r = 0; r < 8; ++r) {
    float mx = -3.0e38f;
#pragma unroll
    for (int j = 0; j < 8; ++j) mx = fmaxf(mx, s[j][r]);
    for (int off = 1; off < 16; off <<= 1) mx = fmaxf(mx, __shfl_xor(mx, off, 32));
    float sum = 0.f;
#pragma unroll
    for (int j = 0; j < 8; ++j) { float p = __expf(s[j][r] - mx); s[j][r] = p; sum += p; }
    for (int off = 1; off < 16; off <<= 1) sum += __shfl_xor(sum, off, 32);
    float inv = 1.f / sum;
    int lr = r + rbase;
#pragma unroll
    for (int j = 0; j < 8; ++j) lds[w][lr][j * 16 + cl] = (__bf16)(s[j][r] * inv);
  }
  __syncthreads();

  // out = attn @ v  (K = 128 keys -> four 32-chunks; N = DH = 64)
  const __bf16* vg = vT_bf + ((size_t)(g * 8 + h) * 64) * 128;
  v8f o[4];
#pragma unroll
  for (int n = 0; n < 4; ++n) o[n] = zero8();
#pragma unroll
  for (int kc = 0; kc < 4; ++kc) {
    v16bf a = load_frag_bf16(&lds[w][0][0], 136, 0, kc * 32);
#pragma unroll
    for (int n = 0; n < 4; ++n) {
      v16bf b = load_frag_bf16(vg, 128, n * 16, kc * 32);
      o[n] = WMMA_BF16(a, b, o[n]);
    }
  }
#pragma unroll
  for (int n = 0; n < 4; ++n) {
#pragma unroll
    for (int r = 0; r < 8; ++r) {
      int node = g * 128 + m0 + r + rbase;
      int col = h * 64 + n * 16 + cl;
      attnout[(size_t)node * 512 + col] = (__bf16)o[n][r];
    }
  }
}

// ---------------------------- fused edge projection (64->512) + scatter-add
// one wave -> 16 edges x all 512 output cols; f32 atomics into scat[T,512]
__global__ void __launch_bounds__(256)
edge_fused(const float* __restrict__ ef, const int* __restrict__ src,
           const __bf16* __restrict__ WeT, float* __restrict__ scat, int E) {
  const int lane = threadIdx.x & 31;
  const int wid  = (int)((blockIdx.x * blockDim.x + threadIdx.x) >> 5);
  const int e0 = wid * 16;
  if (e0 >= E) return;
  v16bf a0 = load_frag_f32(ef, 64, e0, 0);
  v16bf a1 = load_frag_f32(ef, 64, e0, 32);
  const int rbase = (lane & 16) ? 8 : 0;
  const int cl = lane & 15;
  int trow[8];
#pragma unroll
  for (int r = 0; r < 8; ++r) trow[r] = src[e0 + r + rbase];
  for (int nt = 0; nt < 32; ++nt) {
    v16bf b0 = load_frag_bf16(WeT, 64, nt * 16, 0);
    v16bf b1 = load_frag_bf16(WeT, 64, nt * 16, 32);
    v8f acc = zero8();
    acc = WMMA_BF16(a0, b0, acc);
    acc = WMMA_BF16(a1, b1, acc);
    int col = nt * 16 + cl;
#pragma unroll
    for (int r = 0; r < 8; ++r)
      atomicAdd(&scat[(size_t)trow[r] * 512 + col], acc[r]);
  }
}

// ---------------------------------------------------------------- launcher
extern "C" void kernel_launch(void* const* d_in, const int* in_sizes, int n_in,
                              void* d_out, int out_size, void* d_ws, size_t ws_size,
                              hipStream_t stream) {
  (void)in_sizes; (void)n_in; (void)out_size; (void)ws_size;

  const float* node_feature = (const float*)d_in[0];
  const float* edge_feature = (const float*)d_in[1];
  const int*   edge_index   = (const int*)d_in[2];     // [2,E]; src = first E
  const int*   dist_indices = (const int*)d_in[3];
  const float* centrality   = (const float*)d_in[4];
  const float* dist_bias    = (const float*)d_in[5];
  const float* Wq    = (const float*)d_in[6];
  const float* Wkv   = (const float*)d_in[7];
  const float* Wout  = (const float*)d_in[8];
  const float* b_out = (const float*)d_in[9];
  const float* We    = (const float*)d_in[10];
  const float* b_e   = (const float*)d_in[11];
  const float* W1    = (const float*)d_in[12];
  const float* b1    = (const float*)d_in[13];
  const float* prelu = (const float*)d_in[14];
  const float* W2    = (const float*)d_in[15];
  const float* b2    = (const float*)d_in[16];

  constexpr int T = 16384, D = 512, E = 524288;
  const size_t TD = (size_t)T * D;

  char* ws = (char*)d_ws;
  size_t off = 0;
  auto alloc = [&](size_t bytes) -> char* {
    char* p = ws + off;
    off += (bytes + 255) & ~(size_t)255;
    return p;
  };
  int*    deg     = (int*)   alloc((size_t)T * 4);
  float*  invs    = (float*) alloc((size_t)T * 4);
  __bf16* x_bf    = (__bf16*)alloc(TD * 2);
  __bf16* q_bf    = (__bf16*)alloc(TD * 2);
  __bf16* k_bf    = (__bf16*)alloc(TD * 2);
  float*  v_f     = (float*) alloc(TD * 4);
  __bf16* vT_bf   = (__bf16*)alloc(TD * 2);
  __bf16* ao_bf   = (__bf16*)alloc(TD * 2);      // attention output
  float*  dense   = (float*) alloc(TD * 4);
  float*  scat    = (float*) alloc(TD * 4);
  __bf16* h0_bf   = (__bf16*)alloc(TD * 2);
  __bf16* h1_bf   = (__bf16*)alloc(TD * 2);
  __bf16* WqT     = (__bf16*)alloc((size_t)D * D * 2);
  __bf16* WkT     = (__bf16*)alloc((size_t)D * D * 2);
  __bf16* WvT     = (__bf16*)alloc((size_t)D * D * 2);
  __bf16* WoutT   = (__bf16*)alloc((size_t)D * D * 2);
  __bf16* W1T     = (__bf16*)alloc((size_t)D * D * 2);
  __bf16* W2T     = (__bf16*)alloc((size_t)D * D * 2);
  __bf16* WeT     = (__bf16*)alloc((size_t)D * 64 * 2);

  const int TPB = 256;
  const int nTD = (int)(TD / TPB);        // 32768 blocks for T*D elementwise

  // 1. degree
  zero_i32<<<T / TPB, TPB, 0, stream>>>(deg, (size_t)T);
  zero_f32<<<nTD, TPB, 0, stream>>>(scat, TD);
  deg_count<<<E / TPB, TPB, 0, stream>>>(edge_index, deg, E);

  // 2. centrality-encoded node features (bf16) + 1/sqrt(deg)
  node_prep<<<nTD, TPB, 0, stream>>>(node_feature, centrality, deg, x_bf, TD);
  inv_deg_k<<<T / TPB, TPB, 0, stream>>>(deg, invs, T);

  // 3. weights -> bf16 transposed [N][K]
  cvt_transpose<<<(D * D) / TPB, TPB, 0, stream>>>(Wq,   WqT,   D, D, 0,   D,     (size_t)D * D);
  cvt_transpose<<<(D * D) / TPB, TPB, 0, stream>>>(Wkv,  WkT,   D, D, 0,   2 * D, (size_t)D * D);
  cvt_transpose<<<(D * D) / TPB, TPB, 0, stream>>>(Wkv,  WvT,   D, D, D,   2 * D, (size_t)D * D);
  cvt_transpose<<<(D * D) / TPB, TPB, 0, stream>>>(Wout, WoutT, D, D, 0,   D,     (size_t)D * D);
  cvt_transpose<<<(D * D) / TPB, TPB, 0, stream>>>(W1,   W1T,   D, D, 0,   D,     (size_t)D * D);
  cvt_transpose<<<(D * D) / TPB, TPB, 0, stream>>>(W2,   W2T,   D, D, 0,   D,     (size_t)D * D);
  cvt_transpose<<<(D * 64) / TPB, TPB, 0, stream>>>(We,  WeT,  64, D, 0,   D,     (size_t)D * 64);

  // 4. q/k/v projections (WMMA + async-LDS staged weights)
  // blocks = (T/128) * (D/64) = 1024
  const int gemm_blocks = (T / 128) * (D / 64);
  gemm_bf16_kernel<<<gemm_blocks, TPB, 0, stream>>>(x_bf, D, WqT, D, T, D, D,
                                                    nullptr, nullptr, nullptr, q_bf, 1);
  gemm_bf16_kernel<<<gemm_blocks, TPB, 0, stream>>>(x_bf, D, WkT, D, T, D, D,
                                                    nullptr, nullptr, nullptr, k_bf, 1);
  gemm_bf16_kernel<<<gemm_blocks, TPB, 0, stream>>>(x_bf, D, WvT, D, T, D, D,
                                                    nullptr, nullptr, v_f, nullptr, 0);
  pack_vT<<<nTD, TPB, 0, stream>>>(v_f, vT_bf, TD);

  // 5. fused attention: one block per (graph, head)
  attn_kernel<<<128 * 8, TPB, 0, stream>>>(q_bf, k_bf, vT_bf, dist_indices, dist_bias, ao_bf);

  // 6. output projection
  gemm_bf16_kernel<<<gemm_blocks, TPB, 0, stream>>>(ao_bf, D, WoutT, D, T, D, D,
                                                    b_out, nullptr, dense, nullptr, 0);

  // 7. fused edge projection + scatter-add (b_e handled via deg*b_e in combine)
  edge_fused<<<E / (16 * 8), TPB, 0, stream>>>(edge_feature, edge_index, WeT, scat, E);

  // 8. combine + MLP
  combine_k<<<nTD, TPB, 0, stream>>>(dense, scat, deg, invs, b_e, h0_bf, TD);
  gemm_bf16_kernel<<<gemm_blocks, TPB, 0, stream>>>(h0_bf, D, W1T, D, T, D, D,
                                                    b1, prelu, nullptr, h1_bf, 2);
  gemm_bf16_kernel<<<gemm_blocks, TPB, 0, stream>>>(h1_bf, D, W2T, D, T, D, D,
                                                    b2, nullptr, (float*)d_out, nullptr, 0);
}
